// Model_59425167507871
// MI455X (gfx1250) — matrix operand order repaired and checked by reading, not compiled, not run on hardware
//
#include <hip/hip_runtime.h>
#include <hip/hip_bf16.h>
#include <math.h>

// ---------------------------------------------------------------------------
// CDNA5 / gfx1250 fused implementation of:
//   3x (block-interconnect GEMM -> gate16) -> 10-way reduce -> softmax
// GEMMs run on v_wmma_f32_16x16x32_f16 (f16 in, f32 accumulate).
// Hidden activations (16384 x 2600 f32) live only in LDS, never in HBM.
// ---------------------------------------------------------------------------

typedef __attribute__((ext_vector_type(16))) _Float16 v16h;
typedef __attribute__((ext_vector_type(8)))  _Float16 v8h;
typedef __attribute__((ext_vector_type(8)))  float    v8f;

#define BATCH     16384
#define ROWTILE   16
#define NGRID     (BATCH / ROWTILE)       // 1024 workgroups
#define HC_PAD    2604                    // LDS stride (8*HC_PAD % 64 == 32 -> no bank conflict)
#define NGATES    1300
#define APAD      1600                    // padded A layout for layers 1/2: 50 blocks * 32 (K pad)
#define NTHREADS  512                     // 16 wave32 per workgroup (4 per SIMD32)
#define NWAVES    (NTHREADS / 32)

// ---------------------------------------------------------------------------
// gate16 with pre-hoisted softmaxed weights (15 registers, w[i] == ws[i+1])
// ---------------------------------------------------------------------------
struct GateW { float w[15]; };

__device__ __forceinline__ GateW load_gatew(const float* __restrict__ wc /* stride NGATES */) {
    GateW gw;
#pragma unroll
    for (int i = 0; i < 15; ++i) gw.w[i] = wc[(i + 1) * NGATES];
    return gw;
}

__device__ __forceinline__ float gate_eval(float A, float Bv, const GateW& g) {
    const float AB = A * Bv;
    const float S  = A + Bv - AB;
    float v;
    v  = g.w[ 0] * AB;
    v += g.w[ 1] * (A - AB);
    v += g.w[ 2] * A;
    v += g.w[ 3] * (Bv - AB);
    v += g.w[ 4] * Bv;
    v += g.w[ 5] * (S - AB);
    v += g.w[ 6] * S;
    v += g.w[ 7] * (1.f - S);
    v += g.w[ 8] * (1.f - (S - AB));
    v += g.w[ 9] * (1.f - Bv);
    v += g.w[10] * (1.f - (Bv - AB));
    v += g.w[11] * (1.f - A);
    v += g.w[12] * (1.f - (A - AB));
    v += g.w[13] * (1.f - AB);
    v += g.w[14];
    return v;
}

// ---------------------------------------------------------------------------
// x : f32 -> f16 convert (A operand staging for layer 0)
// ---------------------------------------------------------------------------
__global__ void cvt_f32_f16(const float* __restrict__ x, _Float16* __restrict__ y, int n) {
    int i = blockIdx.x * blockDim.x + threadIdx.x;
    if (i < n) y[i] = (_Float16)x[i];
}

// ---------------------------------------------------------------------------
// Interconnect prep: softmax over K (axis=1), f32->f16, store pre-swizzled in
// this code's B-fragment order (contiguous K, as in load_a).  A 32x16 (KxN) f16 B-tile occupies 32 lanes * 16 f16:
//   lanes 0..15  : N = lane,     elements j -> K = j      (K 0..15)
//   lanes 16..31 : N = lane-16,  elements j -> K = 16 + j (K 16..31)
// K < 32 and N beyond NVALID are zero-padded.
// ---------------------------------------------------------------------------
template <int NBLK, int K, int NVALID, int NT>
__global__ void prep_conn(const float* __restrict__ src, _Float16* __restrict__ dst) {
    const int idx = blockIdx.x * blockDim.x + threadIdx.x;
    const int per = NT * 16;
    if (idx >= NBLK * per) return;
    const int n = idx / per;
    const int o = idx % per;

    float ev[32];
#pragma unroll
    for (int k = 0; k < 32; ++k) ev[k] = 0.f;

    float inv = 0.f;
    if (o < NVALID) {
        float mx = -3.4e38f;
#pragma unroll
        for (int k = 0; k < K; ++k) {
            float v = src[((size_t)n * K + k) * NVALID + o];
            mx = fmaxf(mx, v);
        }
        float s = 0.f;
#pragma unroll
        for (int k = 0; k < K; ++k) {
            float e = expf(src[((size_t)n * K + k) * NVALID + o] - mx);
            ev[k] = e;
            s += e;
        }
        inv = 1.f / s;
    }
    const int t    = o >> 4;
    const int lane = o & 15;
    _Float16* d0 = dst + (((size_t)(n * NT + t) * 32) + lane) * 16;        // lanes 0..15
    _Float16* d1 = d0 + 16 * 16;                                           // lanes 16..31
#pragma unroll
    for (int j = 0; j < 16; ++j) {
        d0[j] = (o < NVALID && j      < K) ? (_Float16)(ev[j]      * inv) : (_Float16)0.f;
        d1[j] = (o < NVALID && 16 + j < K) ? (_Float16)(ev[16 + j] * inv) : (_Float16)0.f;
    }
}

// ---------------------------------------------------------------------------
// Gate-weight prep: softmax over axis 0 of (16, 1300)
// ---------------------------------------------------------------------------
__global__ void prep_gatew(const float* __restrict__ w, float* __restrict__ wsk) {
    int g = blockIdx.x * blockDim.x + threadIdx.x;
    if (g >= NGATES) return;
    float mx = -3.4e38f;
#pragma unroll
    for (int k = 0; k < 16; ++k) mx = fmaxf(mx, w[k * NGATES + g]);
    float e[16];
    float s = 0.f;
#pragma unroll
    for (int k = 0; k < 16; ++k) { e[k] = expf(w[k * NGATES + g] - mx); s += e[k]; }
    float inv = 1.f / s;
#pragma unroll
    for (int k = 0; k < 16; ++k) wsk[k * NGATES + g] = e[k] * inv;
}

// ---------------------------------------------------------------------------
// Compile-time N-tile range for one diagonal block: fully unrolled so the
// scheduler can clause the B-fragment b128 loads and keep several WMMAs in
// flight (deep s_wait_loadcnt pipelining), and so the store guard constant-
// folds for full tiles.
// ---------------------------------------------------------------------------
template <int NVALID, int NT, int T0, int T1>
__device__ __forceinline__ void gemm_tiles(int n, const v16h& a,
                                           const _Float16* __restrict__ Bfrag,
                                           float* __restrict__ htile,
                                           int m, int kh, int lane) {
#pragma unroll
    for (int t = T0; t < T1; ++t) {
        // B fragment: pre-swizzled, 16 f16 contiguous per lane (2x b128)
        v16h b = *(const v16h*)(Bfrag + (((size_t)(n * NT + t) * 32) + lane) * 16);
        v8f c = {};
        c = __builtin_amdgcn_wmma_f32_16x16x32_f16(
                /*neg_a=*/false, a, /*neg_b=*/false, b,
                /*c_mod=*/(short)0, c, /*reuse_a=*/false, /*reuse_b=*/false);
        const int cb = t * 16 + m;                    // column within block
        if (cb < NVALID) {                            // folds away for full tiles
            const int col = n * NVALID + cb;
#pragma unroll
            for (int r = 0; r < 8; ++r)
                htile[(r + 8 * kh) * HC_PAD + col] = c[r];
        }
    }
}

// ---------------------------------------------------------------------------
// Fused layer: block-diagonal WMMA GEMM into LDS hidden tile, then gate16.
//   NBLK   : number of diagonal blocks (8 or 50)
//   NVALID : valid output columns per block (325 or 52)
//   NT     : 16-wide N tiles per block (21 or 4)
//   LDA    : row stride (f16) of the A activation buffer (256 or 1600)
//   FINAL  : last layer -> gate + 10-way reduce + softmax to d_out
// One workgroup (16 wave32) per 16-row tile.
//   NBLK >= NWAVES : wave w owns blocks n = w, w+NWAVES, ...
//   NBLK <  NWAVES : two waves split a block into contiguous tile ranges.
// ---------------------------------------------------------------------------
template <int NBLK, int NVALID, int NT, int LDA, bool FINAL>
__global__ void fused_layer(const _Float16* __restrict__ Ain,
                            const _Float16* __restrict__ Bfrag,
                            const float*    __restrict__ wsk,
                            _Float16*       __restrict__ gateOut,
                            float*          __restrict__ finalOut) {
    extern __shared__ float smem[];
    float* htile = smem;                              // 16 * HC_PAD f32

    const int tid  = threadIdx.x;
    const int wave = tid >> 5;
    const int lane = tid & 31;
    const int row0 = blockIdx.x * ROWTILE;
    const int m    = lane & 15;                       // A row / D column within tile
    const int kh   = lane >> 4;                       // K-half select (ISA A/D layout)

    // A fragment loader: 16x32 f16, lane m = row, one contiguous 16-f16 run (K = 16*kh + i).
    auto load_a = [&](int n) -> v16h {
        const _Float16* pA = Ain + (size_t)(row0 + m) * LDA + n * 32;
        v16h a = *(const v16h*)(pA + kh * 16);
        return a;
    };

    // ---- GEMM phase: one v_wmma_f32_16x16x32_f16 per (block, N-tile) ----
    if constexpr (NBLK >= NWAVES) {
        for (int n = wave; n < NBLK; n += NWAVES) {
            const v16h a = load_a(n);
            gemm_tiles<NVALID, NT, 0, NT>(n, a, Bfrag, htile, m, kh, lane);
        }
    } else {
        // 8 blocks, 16 waves: two waves per block, contiguous compile-time halves
        constexpr int TH = (NT + 1) / 2;
        const int n = wave & (NBLK - 1);
        const v16h a = load_a(n);
        if (wave < NBLK) gemm_tiles<NVALID, NT, 0,  TH>(n, a, Bfrag, htile, m, kh, lane);
        else             gemm_tiles<NVALID, NT, TH, NT>(n, a, Bfrag, htile, m, kh, lane);
    }
    __syncthreads();

    if constexpr (!FINAL) {
        // ---- Gate phase: weights hoisted per gate, reused across all 16 rows.
        //      Output f16, padded to next layer's (50 x 32) A layout. ----
        for (int p = tid; p < APAD; p += NTHREADS) {
            const int k   = p & 31;
            const int blk = p >> 5;
            if (k < 26) {
                const int g = blk * 26 + k;
                const GateW gw = load_gatew(wsk + g);
#pragma unroll
                for (int r = 0; r < ROWTILE; ++r) {
                    const float* hrow = htile + r * HC_PAD;
                    const float val = gate_eval(hrow[2 * g], hrow[2 * g + 1], gw);
                    gateOut[(size_t)(row0 + r) * APAD + p] = (_Float16)val;
                }
            } else {
#pragma unroll
                for (int r = 0; r < ROWTILE; ++r)
                    gateOut[(size_t)(row0 + r) * APAD + p] = (_Float16)0.f;
            }
        }
    } else {
        // ---- Final phase: gate -> 10 logits -> softmax (deterministic) ----
        float* gv = smem + ROWTILE * HC_PAD;          // 16 * 1300 gate values
        float* lg = gv + ROWTILE * NGATES;            // 16 * 10 logits
        for (int g = tid; g < NGATES; g += NTHREADS) {
            const GateW gw = load_gatew(wsk + g);
#pragma unroll
            for (int r = 0; r < ROWTILE; ++r) {
                const float* hrow = htile + r * HC_PAD;
                gv[r * NGATES + g] = gate_eval(hrow[2 * g], hrow[2 * g + 1], gw);
            }
        }
        __syncthreads();
        if (tid < 160) {                              // one thread per (row, category)
            const int r = tid / 10;
            const int j = tid - r * 10;
            const float* base = gv + r * NGATES + j * 130;
            float s = 0.f;
            for (int i = 0; i < 130; ++i) s += base[i];
            lg[tid] = s;
        }
        __syncthreads();
        if (tid < ROWTILE) {
            const int r = tid;
            float mx = -3.4e38f;
#pragma unroll
            for (int j = 0; j < 10; ++j) mx = fmaxf(mx, lg[r * 10 + j]);
            float e[10];
            float s = 0.f;
#pragma unroll
            for (int j = 0; j < 10; ++j) { e[j] = expf(lg[r * 10 + j] - mx); s += e[j]; }
            const float inv = 1.f / s;
#pragma unroll
            for (int j = 0; j < 10; ++j)
                finalOut[(size_t)(row0 + r) * 10 + j] = e[j] * inv;
        }
    }
}

// ---------------------------------------------------------------------------
// Host side
// ---------------------------------------------------------------------------
extern "C" void kernel_launch(void* const* d_in, const int* in_sizes, int n_in,
                              void* d_out, int out_size, void* d_ws, size_t ws_size,
                              hipStream_t stream) {
    const float* x  = (const float*)d_in[0];   // (16384, 256)
    const float* c0 = (const float*)d_in[1];   // (8, 32, 325)
    const float* w0 = (const float*)d_in[2];   // (16, 1300)
    const float* c1 = (const float*)d_in[3];   // (50, 26, 52)
    const float* w1 = (const float*)d_in[4];   // (16, 1300)
    const float* c2 = (const float*)d_in[5];   // (50, 26, 52)
    const float* w2 = (const float*)d_in[6];   // (16, 1300)
    float* out = (float*)d_out;                // (16384, 10)

    // workspace carve-up (all chunk sizes are multiples of 256 bytes)
    char* ws = (char*)d_ws;
    size_t off = 0;
    auto carve = [&](size_t bytes) { char* p = ws + off; off += bytes; return p; };
    _Float16* xh     = (_Float16*)carve((size_t)BATCH * 256 * 2);          //  8.39 MB
    _Float16* conn0f = (_Float16*)carve((size_t)8  * 21 * 32 * 16 * 2);    //  0.17 MB
    _Float16* conn1f = (_Float16*)carve((size_t)50 * 4  * 32 * 16 * 2);    //  0.20 MB
    _Float16* conn2f = (_Float16*)carve((size_t)50 * 4  * 32 * 16 * 2);    //  0.20 MB
    float*    ws0    = (float*)   carve((size_t)16 * NGATES * 4);          //  0.08 MB
    float*    ws1    = (float*)   carve((size_t)16 * NGATES * 4);
    float*    ws2    = (float*)   carve((size_t)16 * NGATES * 4);
    _Float16* g0p    = (_Float16*)carve((size_t)BATCH * APAD * 2);         // 52.43 MB
    _Float16* g1p    = (_Float16*)carve((size_t)BATCH * APAD * 2);         // 52.43 MB
    (void)ws_size; (void)in_sizes; (void)n_in; (void)out_size;

    // ---- prep ----
    {
        const int n = BATCH * 256;
        cvt_f32_f16<<<(n + 255) / 256, 256, 0, stream>>>(x, xh, n);
    }
    prep_conn<8, 32, 325, 21><<<(8 * 21 * 16 + 255) / 256, 256, 0, stream>>>(c0, conn0f);
    prep_conn<50, 26, 52, 4><<<(50 * 4 * 16 + 255) / 256, 256, 0, stream>>>(c1, conn1f);
    prep_conn<50, 26, 52, 4><<<(50 * 4 * 16 + 255) / 256, 256, 0, stream>>>(c2, conn2f);
    prep_gatew<<<(NGATES + 255) / 256, 256, 0, stream>>>(w0, ws0);
    prep_gatew<<<(NGATES + 255) / 256, 256, 0, stream>>>(w1, ws1);
    prep_gatew<<<(NGATES + 255) / 256, 256, 0, stream>>>(w2, ws2);

    // ---- fused layers ----
    const size_t lds_gemm  = (size_t)ROWTILE * HC_PAD * sizeof(float);                 // 166,656 B
    const size_t lds_final = lds_gemm + (size_t)ROWTILE * NGATES * sizeof(float)
                                      + (size_t)ROWTILE * 10 * sizeof(float);          // 250,496 B

    fused_layer<8, 325, 21, 256,  false><<<NGRID, NTHREADS, lds_gemm,  stream>>>(
        xh,  conn0f, ws0, g0p, nullptr);
    fused_layer<50, 52, 4,  APAD, false><<<NGRID, NTHREADS, lds_gemm,  stream>>>(
        g0p, conn1f, ws1, g1p, nullptr);
    fused_layer<50, 52, 4,  APAD, true ><<<NGRID, NTHREADS, lds_final, stream>>>(
        g1p, conn2f, ws2, nullptr, out);
}
